// NSVQ_23132693856375
// MI455X (gfx1250) — compile-verified, matrix operand order checked by reading, not run on hardware
//
#include <hip/hip_runtime.h>

// ---------- types ----------
typedef __attribute__((ext_vector_type(16))) __bf16 v16bf;
typedef __attribute__((ext_vector_type(8)))  float  v8f;
typedef __attribute__((ext_vector_type(4)))  unsigned int u32x4;
typedef int v4i __attribute__((vector_size(16)));   // matches async-LDS builtin param

union FragB16 {
    v16bf v;
    u32x4 q[2];
    unsigned short h[16];
};

// native f32 -> bf16 (RNE) — lets the backend pick v_cvt_*bf16_f32
__device__ __forceinline__ unsigned short f2bf(float f) {
    union { __bf16 b; unsigned short s; } u;
    u.b = (__bf16)f;
    return u.s;
}

// ---------- problem constants ----------
static constexpr int DIMC = 256;   // channel dim
static constexpr int TT   = 2048;  // time
static constexpr int NN   = 32768; // B*T rows
static constexpr int KK   = 1024;  // codebook entries
static constexpr int DD   = 64;    // embedding dim

// ---------- async global->LDS staging of a 32KB block with 256 threads ----
// 16384 ushorts = 8 x b128 per thread. Uses the CDNA5 async-to-LDS path
// (ASYNCcnt) when the builtin exists; falls back to plain LDS stores.
__device__ __forceinline__ void stage32k(const unsigned short* __restrict__ g,
                                         unsigned short* l, int tid) {
#if defined(__has_builtin) && __has_builtin(__builtin_amdgcn_global_load_async_to_lds_b128)
#pragma unroll
    for (int i = 0; i < 8; ++i) {
        const int off = (tid + i * 256) * 8;      // ushort index, 16B granules
        __builtin_amdgcn_global_load_async_to_lds_b128(
            (__attribute__((address_space(1))) v4i*)(g + off),
            (__attribute__((address_space(3))) v4i*)(l + off),
            0, 0);
    }
#if defined(__has_builtin) && __has_builtin(__builtin_amdgcn_s_wait_asynccnt)
    __builtin_amdgcn_s_wait_asynccnt(0);
#else
    asm volatile("s_wait_asynccnt 0" ::: "memory");
#endif
#else
#pragma unroll
    for (int i = 0; i < 8; ++i) {
        const int off = (tid + i * 256) * 8;
        *(u32x4*)(l + off) = *(const u32x4*)(g + off);
    }
#endif
}

// =====================================================================
// Kernel 1: precompute bf16 weights (transposed), bf16 codebooks, ||c||^2
// =====================================================================
__global__ void k_setup(const float* __restrict__ codebooks,
                        const float* __restrict__ W_in,
                        const float* __restrict__ W_out,
                        unsigned short* __restrict__ cb_bf,   // [K, D]
                        unsigned short* __restrict__ WinT,    // [D, DIM]
                        unsigned short* __restrict__ WoutT,   // [DIM, D]
                        float* __restrict__ cnorm)            // [K]
{
    int tid = blockIdx.x * blockDim.x + threadIdx.x;   // 65536 threads
    if (tid < KK * DD) cb_bf[tid] = f2bf(codebooks[tid]);
    if (tid < DD * DIMC) {               // WinT[d][c] = W_in[c][d]
        int d = tid / DIMC, c = tid % DIMC;
        WinT[tid] = f2bf(W_in[c * DD + d]);
    }
    if (tid < DIMC * DD) {               // WoutT[ch][d] = W_out[d][ch]
        int ch = tid / DD, dd = tid % DD;
        WoutT[tid] = f2bf(W_out[dd * DIMC + ch]);
    }
    if (tid < KK) {
        const float* c = codebooks + (size_t)tid * DD;
        float s = 0.f;
        for (int i = 0; i < DD; ++i) s += c[i] * c[i];
        cnorm[tid] = s;
    }
}

// =====================================================================
// Kernel 2: encode  x[m,d] = sum_c input[b,c,t] * W_in[c,d] + b_in[d]
// one wave = 16 rows x 64 cols, K-loop over 256 -> 32 WMMAs/wave
// W_in^T staged once per block in LDS (async), B-frags read via ds_load
// =====================================================================
__global__ void k_encode(const float* __restrict__ in,              // [B, DIM, T]
                         const unsigned short* __restrict__ WinT,   // [D, DIM] bf16
                         const float* __restrict__ b_in,            // [D]
                         float* __restrict__ x_f32,                 // [N, D]
                         unsigned short* __restrict__ x_bf)         // [N, D] bf16
{
    __shared__ unsigned short wlds[DD * DIMC];   // 32 KB
    stage32k(WinT, wlds, threadIdx.x);
    __syncthreads();

    const int lane = threadIdx.x & 31;
    const int wid  = threadIdx.x >> 5;
    const int l16  = lane & 15;
    const int hi   = lane >> 4;
    const int row_base = (blockIdx.x * 8 + wid) * 16;
    const int m = row_base + l16;
    const int b = m >> 11;             // T = 2048
    const int t = m & 2047;
    const float* inb = in + (size_t)b * DIMC * TT + t;

    v8f acc[4];
#pragma unroll
    for (int ct = 0; ct < 4; ++ct)
#pragma unroll
        for (int r = 0; r < 8; ++r) acc[ct][r] = 0.f;

    for (int kc = 0; kc < 8; ++kc) {          // c tile of 32
        const int cb0 = kc * 32;
        FragB16 a;
#pragma unroll
        for (int e = 0; e < 8; ++e) {
            const int c0 = cb0 + hi * 8 + e;
            const int c1 = cb0 + 16 + hi * 8 + e;
            a.h[e]     = f2bf(inb[(size_t)c0 * TT]);
            a.h[e + 8] = f2bf(inb[(size_t)c1 * TT]);
        }
#pragma unroll
        for (int ct = 0; ct < 4; ++ct) {
            const int d = ct * 16 + l16;
            const u32x4* wrow = (const u32x4*)(wlds + (size_t)d * DIMC);
            FragB16 bf;
            bf.q[0] = wrow[(cb0 >> 3) + hi];
            bf.q[1] = wrow[(cb0 >> 3) + 2 + hi];
            acc[ct] = __builtin_amdgcn_wmma_f32_16x16x32_bf16(
                false, a.v, false, bf.v, (short)0, acc[ct], false, false);
        }
    }
#pragma unroll
    for (int ct = 0; ct < 4; ++ct) {
        const int d = ct * 16 + l16;
        const float bias = b_in[d];
#pragma unroll
        for (int r = 0; r < 8; ++r) {
            const int row = row_base + r + 8 * hi;
            const float v = acc[ct][r] + bias;
            x_f32[(size_t)row * DD + d] = v;
            x_bf [(size_t)row * DD + d] = f2bf(v);
        }
    }
}

// =====================================================================
// Kernel 3: distances + argmin.  argmin_k( ||c_k||^2 - 2 x.c_k )
// one wave = 16 rows vs all 1024 codebooks -> 128 WMMAs/wave
// codebook staged in LDS in 4 x 32KB chunks (async), shared by 8 waves
// =====================================================================
__global__ void k_dist(const unsigned short* __restrict__ x_bf,   // [N, D]
                       const unsigned short* __restrict__ cb_bf,  // [K, D]
                       const float* __restrict__ cnorm,           // [K]
                       int* __restrict__ midx)                    // [N]
{
    __shared__ unsigned short cbs[256 * DD];   // 32 KB chunk (256 codebooks)
    __shared__ float cns[256];

    const int lane = threadIdx.x & 31;
    const int wid  = threadIdx.x >> 5;
    const int l16  = lane & 15;
    const int hi   = lane >> 4;
    const int row_base = (blockIdx.x * 8 + wid) * 16;

    const u32x4* xr = (const u32x4*)(x_bf + (size_t)(row_base + l16) * DD);
    FragB16 a0, a1;
    a0.q[0] = xr[hi];     a0.q[1] = xr[2 + hi];
    a1.q[0] = xr[4 + hi]; a1.q[1] = xr[6 + hi];

    float mval[8];
    int   mi[8];
#pragma unroll
    for (int r = 0; r < 8; ++r) { mval[r] = 3.4e38f; mi[r] = 0; }

    for (int chunk = 0; chunk < 4; ++chunk) {
        stage32k(cb_bf + chunk * 256 * DD, cbs, threadIdx.x);
        cns[threadIdx.x] = cnorm[chunk * 256 + threadIdx.x];
        __syncthreads();

        for (int kt = 0; kt < 16; ++kt) {
            const int nl = kt * 16 + l16;            // codebook within chunk
            const int n  = chunk * 256 + nl;         // global codebook index
            const u32x4* cr = (const u32x4*)(cbs + (size_t)nl * DD);
            FragB16 b0, b1;
            b0.q[0] = cr[hi];     b0.q[1] = cr[2 + hi];
            b1.q[0] = cr[4 + hi]; b1.q[1] = cr[6 + hi];
            v8f acc;
#pragma unroll
            for (int r = 0; r < 8; ++r) acc[r] = 0.f;
            acc = __builtin_amdgcn_wmma_f32_16x16x32_bf16(false, a0.v, false, b0.v,
                                                          (short)0, acc, false, false);
            acc = __builtin_amdgcn_wmma_f32_16x16x32_bf16(false, a1.v, false, b1.v,
                                                          (short)0, acc, false, false);
            const float cn = cns[nl];
#pragma unroll
            for (int r = 0; r < 8; ++r) {
                const float dist = cn - 2.0f * acc[r];
                if (dist < mval[r]) { mval[r] = dist; mi[r] = n; }
            }
        }
        __syncthreads();
    }
    // reduce over the 16 lanes holding the same row (xor<=8 stays in group)
#pragma unroll
    for (int r = 0; r < 8; ++r) {
#pragma unroll
        for (int off = 8; off >= 1; off >>= 1) {
            const float ov = __shfl_xor(mval[r], off, 32);
            const int   oi = __shfl_xor(mi[r],   off, 32);
            if (ov < mval[r] || (ov == mval[r] && oi < mi[r])) {
                mval[r] = ov; mi[r] = oi;
            }
        }
        if (l16 == 0) midx[row_base + r + 8 * hi] = mi[r];
    }
}

// =====================================================================
// Kernel 4: noise substitution.  q = x + (||x-hard|| / (||rv||+eps)) * rv
// one wave per row (lane handles d=lane, d=lane+32)
// =====================================================================
__global__ void k_vq(const float* __restrict__ x_f32,      // [N, D]
                     const float* __restrict__ codebooks,  // [K, D] f32
                     const float* __restrict__ rv,         // [N, D]
                     const int* __restrict__ midx,         // [N]
                     unsigned short* __restrict__ q_bf)    // [N, D] bf16
{
    const int lane = threadIdx.x & 31;
    const int wid  = threadIdx.x >> 5;
    const int row  = blockIdx.x * 8 + wid;
    const int idx  = midx[row];
    const float* xr = x_f32     + (size_t)row * DD;
    const float* cr = codebooks + (size_t)idx * DD;
    const float* rr = rv        + (size_t)row * DD;
    const float x0 = xr[lane], x1 = xr[lane + 32];
    const float c0 = cr[lane], c1 = cr[lane + 32];
    const float r0 = rr[lane], r1 = rr[lane + 32];
    const float d0 = x0 - c0, d1 = x1 - c1;
    float sres = d0 * d0 + d1 * d1;
    float srnd = r0 * r0 + r1 * r1;
#pragma unroll
    for (int off = 16; off >= 1; off >>= 1) {
        sres += __shfl_xor(sres, off, 32);
        srnd += __shfl_xor(srnd, off, 32);
    }
    const float scale = sqrtf(sres) / (sqrtf(srnd) + 1e-12f);
    q_bf[(size_t)row * DD + lane]      = f2bf(x0 + scale * r0);
    q_bf[(size_t)row * DD + lane + 32] = f2bf(x1 + scale * r1);
}

// =====================================================================
// Kernel 5: decode  out[b,ch,t] = sum_d q[m,d]*W_out[d,ch] + b_out[ch]
// one wave = 16 rows x 256 cols (16 tiles x 2 WMMAs), W_out^T staged in
// LDS (async), LDS transpose for contiguous [B,DIM,T] stores
// =====================================================================
__global__ void k_decode(const unsigned short* __restrict__ q_bf,   // [N, D]
                         const unsigned short* __restrict__ WoutT,  // [DIM, D] bf16
                         const float* __restrict__ b_out,           // [DIM]
                         float* __restrict__ out)                   // [B, DIM, T]
{
    __shared__ unsigned short wlds[DIMC * DD];   // 32 KB
    __shared__ float lds[8 * 16 * 17];           // per-wave transpose tiles
    stage32k(WoutT, wlds, threadIdx.x);
    __syncthreads();

    const int lane = threadIdx.x & 31;
    const int wid  = threadIdx.x >> 5;
    const int l16  = lane & 15;
    const int hi   = lane >> 4;
    const int row_base = (blockIdx.x * 8 + wid) * 16;
    const int b = row_base >> 11;
    const int t_base = row_base & 2047;
    float* tile = lds + wid * 16 * 17;

    const u32x4* qr = (const u32x4*)(q_bf + (size_t)(row_base + l16) * DD);
    FragB16 a0, a1;
    a0.q[0] = qr[hi];     a0.q[1] = qr[2 + hi];
    a1.q[0] = qr[4 + hi]; a1.q[1] = qr[6 + hi];

    const int rdch = lane >> 1;        // ch within tile for readback
    const int toff = (lane & 1) * 8;   // t sub-offset

    for (int ct = 0; ct < 16; ++ct) {
        const int ch = ct * 16 + l16;
        const u32x4* wr = (const u32x4*)(wlds + (size_t)ch * DD);
        FragB16 b0, b1;
        b0.q[0] = wr[hi];     b0.q[1] = wr[2 + hi];
        b1.q[0] = wr[4 + hi]; b1.q[1] = wr[6 + hi];
        v8f acc;
#pragma unroll
        for (int r = 0; r < 8; ++r) acc[r] = 0.f;
        acc = __builtin_amdgcn_wmma_f32_16x16x32_bf16(false, a0.v, false, b0.v,
                                                      (short)0, acc, false, false);
        acc = __builtin_amdgcn_wmma_f32_16x16x32_bf16(false, a1.v, false, b1.v,
                                                      (short)0, acc, false, false);
        const float bias = b_out[ch];
#pragma unroll
        for (int r = 0; r < 8; ++r)
            tile[l16 * 17 + r + 8 * hi] = acc[r] + bias;   // tile[ch_loc][t_loc]
        __syncthreads();
        // each lane writes 8 contiguous t for one channel (2x b128 stores)
        const int och = ct * 16 + rdch;
        float4 v0, v1;
        const float* trow = tile + rdch * 17 + toff;
        v0.x = trow[0]; v0.y = trow[1]; v0.z = trow[2]; v0.w = trow[3];
        v1.x = trow[4]; v1.y = trow[5]; v1.z = trow[6]; v1.w = trow[7];
        float* op = out + ((size_t)b * DIMC + och) * TT + t_base + toff;
        *(float4*)op       = v0;
        *(float4*)(op + 4) = v1;
        __syncthreads();
    }
}

// =====================================================================
extern "C" void kernel_launch(void* const* d_in, const int* in_sizes, int n_in,
                              void* d_out, int out_size, void* d_ws, size_t ws_size,
                              hipStream_t stream) {
    (void)in_sizes; (void)n_in; (void)out_size; (void)ws_size;
    const float* input     = (const float*)d_in[0];   // [16, 256, 2048]
    const float* codebooks = (const float*)d_in[1];   // [1024, 64]
    const float* W_in      = (const float*)d_in[2];   // [256, 64]
    const float* b_in      = (const float*)d_in[3];   // [64]
    const float* W_out     = (const float*)d_in[4];   // [64, 256]
    const float* b_out     = (const float*)d_in[5];   // [256]
    const float* rv        = (const float*)d_in[6];   // [32768, 64]
    float* out = (float*)d_out;

    // workspace layout (all offsets 256B aligned)
    char* ws = (char*)d_ws;
    size_t off = 0;
    float*          x_f32 = (float*)(ws + off);          off += (size_t)NN * DD * 4;   // 8 MB
    unsigned short* x_bf  = (unsigned short*)(ws + off); off += (size_t)NN * DD * 2;   // 4 MB
    unsigned short* q_bf  = (unsigned short*)(ws + off); off += (size_t)NN * DD * 2;   // 4 MB
    int*            midx  = (int*)(ws + off);            off += (size_t)NN * 4;        // 128 KB
    unsigned short* WinT  = (unsigned short*)(ws + off); off += (size_t)DD * DIMC * 2; // 32 KB
    unsigned short* WoutT = (unsigned short*)(ws + off); off += (size_t)DIMC * DD * 2; // 32 KB
    unsigned short* cb_bf = (unsigned short*)(ws + off); off += (size_t)KK * DD * 2;   // 128 KB
    float*          cnorm = (float*)(ws + off);          off += (size_t)KK * 4;        // 4 KB

    k_setup <<<256, 256, 0, stream>>>(codebooks, W_in, W_out, cb_bf, WinT, WoutT, cnorm);
    k_encode<<<256, 256, 0, stream>>>(input, WinT, b_in, x_f32, x_bf);    // 256*8 waves * 16 rows
    k_dist  <<<256, 256, 0, stream>>>(x_bf, cb_bf, cnorm, midx);
    k_vq    <<<4096, 256, 0, stream>>>(x_f32, codebooks, rv, midx, q_bf); // 4096*8 rows
    k_decode<<<256, 256, 0, stream>>>(q_bf, WoutT, b_out, out);
}
